// GatingNetwork_91027536872044
// MI455X (gfx1250) — matrix operand order repaired and checked by reading, not compiled, not run on hardware
//
#include <hip/hip_runtime.h>

// ---------------------------------------------------------------------------
// MoE gating network for MI455X (gfx1250, wave32, WMMA)
//   B=4, S=8192, D=1024, E=64, K=2, capacity=640
// GEMM (32768x1024 @ 1024x64) via v_wmma_f32_16x16x32_f16 with W_gate staged
// as fp16 in LDS; fused softmax + top-2 via ds_swizzle XOR butterflies;
// sequential capacity mask via blocked exclusive scan.
// ---------------------------------------------------------------------------

typedef __attribute__((ext_vector_type(16))) _Float16 v16h;
typedef __attribute__((ext_vector_type(8)))  _Float16 v8h;
typedef __attribute__((ext_vector_type(4)))  _Float16 v4h;
typedef __attribute__((ext_vector_type(8)))  float    v8f;

#define DD 1024            // hidden dim
#define EE 64              // experts
#define NTOK 32768         // B*S
#define NSLOT 65536        // NTOK*K
#define CAPACITY 640       // int(NTOK/E * 1.25)
#define LDSB 1032          // padded LDS row stride (halves) for W_gate
#define HBLK 256           // scan blocks
#define SLOTS_PER_BLK 256  // slots per scan block (128 tokens, K-aligned)

// XOR-lane exchange via ds_swizzle group-of-32 mode (single DS op, no index
// math). offset = {xor_mask[14:10], or_mask[9:5]=0, and_mask[4:0]=0x1f}.
// Masks 1/2/4/8 never cross a 16-lane half-wave, which is the reduction
// domain dictated by the WMMA C-layout.
template <int XORM>
__device__ __forceinline__ float swz_xor_f(float v) {
  return __int_as_float(
      __builtin_amdgcn_ds_swizzle(__float_as_int(v), (XORM << 10) | 0x1f));
}
template <int XORM>
__device__ __forceinline__ int swz_xor_i(int v) {
  return __builtin_amdgcn_ds_swizzle(v, (XORM << 10) | 0x1f);
}

__device__ __forceinline__ void topk_ins(float& p1, int& c1, float& p2, int& c2,
                                         float q, int c) {
  if (q > p1 || (q == p1 && c < c1)) {
    p2 = p1; c2 = c1; p1 = q; c1 = c;
  } else if (q > p2 || (q == p2 && c < c2)) {
    p2 = q; c2 = c;
  }
}

// ---------------------------------------------------------------------------
// Kernel 0: zero the per-expert prob-sum accumulators (ws is poisoned).
// ---------------------------------------------------------------------------
__global__ void moe_init(float* __restrict__ expert_sum) {
  expert_sum[threadIdx.x] = 0.0f;
}

// ---------------------------------------------------------------------------
// Kernel 1: GEMM + softmax + top-2.
// Grid: 256 blocks x 256 threads (8 waves). Block covers 128 token rows.
// Each wave: 16 M rows x 64 N cols (4 WMMA accumulators), K=1024 in 32 steps.
// ---------------------------------------------------------------------------
__global__ void __launch_bounds__(256)
moe_gate_gemm(const float* __restrict__ hs,       // [NTOK, DD]
              const float* __restrict__ wg,       // [EE, DD]
              float* __restrict__ out_gate,       // [NTOK, EE] softmax probs
              float* __restrict__ ws_prob,        // [NSLOT] top-k probs
              int*   __restrict__ ws_idx,         // [NSLOT] top-k expert ids
              float* __restrict__ expert_sum) {   // [EE] sum of probs
  extern __shared__ char smem[];
  float*    smSums = (float*)smem;                 // 64 floats
  _Float16* Bs     = (_Float16*)(smem + 256);      // [EE][LDSB] halves

  const int tid = threadIdx.x;
  if (tid < EE) smSums[tid] = 0.0f;

  // Stage W_gate (fp32 -> fp16) into LDS; 16384 float4 segments.
  for (int i = tid; i < (EE * DD) / 4; i += 256) {
    int flat = i * 4;
    int e = flat >> 10;          // / DD
    int k = flat & (DD - 1);
    float4 f = *(const float4*)(wg + flat);
    v4h h;
    h.x = (_Float16)f.x; h.y = (_Float16)f.y;
    h.z = (_Float16)f.z; h.w = (_Float16)f.w;
    *(v4h*)&Bs[e * LDSB + k] = h;
  }
  __syncthreads();

  const int wave = tid >> 5;
  const int lane = tid & 31;
  const int lo   = lane & 15;
  const int hi   = lane >> 4;
  const int m0   = blockIdx.x * 128 + wave * 16;

  v8f acc0 = {}, acc1 = {}, acc2 = {}, acc3 = {};
  const float* arow = hs + (size_t)(m0 + lo) * DD;

#pragma unroll 1
  for (int kk = 0; kk < DD / 32; ++kk) {
    const int kbase = kk * 32;
    // A fragment (16x32 f16): lane row = lo; halves 0..15 -> K=kbase+hi*16..+15.
    float4 f0 = *(const float4*)(arow + kbase + hi * 16);
    float4 f1 = *(const float4*)(arow + kbase + hi * 16 + 4);
    float4 f2 = *(const float4*)(arow + kbase + hi * 16 + 8);
    float4 f3 = *(const float4*)(arow + kbase + hi * 16 + 12);
    // Speculative prefetch of next k-chunk (dropped silently if OOB).
    __builtin_prefetch(arow + kbase + 32, 0, 3);

    v16h a;
    a[0]  = (_Float16)f0.x; a[1]  = (_Float16)f0.y;
    a[2]  = (_Float16)f0.z; a[3]  = (_Float16)f0.w;
    a[4]  = (_Float16)f1.x; a[5]  = (_Float16)f1.y;
    a[6]  = (_Float16)f1.z; a[7]  = (_Float16)f1.w;
    a[8]  = (_Float16)f2.x; a[9]  = (_Float16)f2.y;
    a[10] = (_Float16)f2.z; a[11] = (_Float16)f2.w;
    a[12] = (_Float16)f3.x; a[13] = (_Float16)f3.y;
    a[14] = (_Float16)f3.z; a[15] = (_Float16)f3.w;

    // B fragments (32x16 f16) from LDS: lane col = lo, K = kbase+hi*16..+15.
    const _Float16* bb = Bs + (size_t)lo * LDSB + kbase + hi * 16;
    union HU { v16h v; v8h h[2]; };
    HU b0, b1, b2, b3;
    b0.h[0] = *(const v8h*)(bb);
    b0.h[1] = *(const v8h*)(bb + 8);
    b1.h[0] = *(const v8h*)(bb + 16 * LDSB);
    b1.h[1] = *(const v8h*)(bb + 16 * LDSB + 8);
    b2.h[0] = *(const v8h*)(bb + 32 * LDSB);
    b2.h[1] = *(const v8h*)(bb + 32 * LDSB + 8);
    b3.h[0] = *(const v8h*)(bb + 48 * LDSB);
    b3.h[1] = *(const v8h*)(bb + 48 * LDSB + 8);

    acc0 = __builtin_amdgcn_wmma_f32_16x16x32_f16(false, a, false, b0.v,
                                                  (short)0, acc0, false, false);
    acc1 = __builtin_amdgcn_wmma_f32_16x16x32_f16(false, a, false, b1.v,
                                                  (short)0, acc1, false, false);
    acc2 = __builtin_amdgcn_wmma_f32_16x16x32_f16(false, a, false, b2.v,
                                                  (short)0, acc2, false, false);
    acc3 = __builtin_amdgcn_wmma_f32_16x16x32_f16(false, a, false, b3.v,
                                                  (short)0, acc3, false, false);
  }

  // C/D layout: lane holds rows m0 + r + 8*hi (r=0..7), col = lo + 16*t.
  // Row's 64 probs live in the 16 lanes of one half-wave -> XOR butterflies
  // with masks 1/2/4/8 (stay within the half-wave).
  float ps0 = 0.f, ps1 = 0.f, ps2 = 0.f, ps3 = 0.f;
#pragma unroll
  for (int r = 0; r < 8; ++r) {
    float v0 = acc0[r], v1 = acc1[r], v2 = acc2[r], v3 = acc3[r];
    float mx = fmaxf(fmaxf(v0, v1), fmaxf(v2, v3));
    mx = fmaxf(mx, swz_xor_f<1>(mx));
    mx = fmaxf(mx, swz_xor_f<2>(mx));
    mx = fmaxf(mx, swz_xor_f<4>(mx));
    mx = fmaxf(mx, swz_xor_f<8>(mx));
    float e0 = __expf(v0 - mx), e1 = __expf(v1 - mx);
    float e2 = __expf(v2 - mx), e3 = __expf(v3 - mx);
    float s = e0 + e1 + e2 + e3;
    s += swz_xor_f<1>(s);
    s += swz_xor_f<2>(s);
    s += swz_xor_f<4>(s);
    s += swz_xor_f<8>(s);
    float inv = 1.0f / s;
    float p0 = e0 * inv, p1 = e1 * inv, p2 = e2 * inv, p3 = e3 * inv;

    const int row = m0 + r + 8 * hi;
    float* g = out_gate + (size_t)row * EE + lo;
    g[0] = p0; g[16] = p1; g[32] = p2; g[48] = p3;
    ps0 += p0; ps1 += p1; ps2 += p2; ps3 += p3;

    // top-2 over 64 cols (butterfly merge of sorted pairs)
    float t1 = -1e30f, t2 = -1e30f;
    int c1 = -1, c2 = -1;
    topk_ins(t1, c1, t2, c2, p0, lo);
    topk_ins(t1, c1, t2, c2, p1, lo + 16);
    topk_ins(t1, c1, t2, c2, p2, lo + 32);
    topk_ins(t1, c1, t2, c2, p3, lo + 48);

    { // off = 1
      float o1 = swz_xor_f<1>(t1); int q1 = swz_xor_i<1>(c1);
      float o2 = swz_xor_f<1>(t2); int q2 = swz_xor_i<1>(c2);
      topk_ins(t1, c1, t2, c2, o1, q1);
      topk_ins(t1, c1, t2, c2, o2, q2);
    }
    { // off = 2
      float o1 = swz_xor_f<2>(t1); int q1 = swz_xor_i<2>(c1);
      float o2 = swz_xor_f<2>(t2); int q2 = swz_xor_i<2>(c2);
      topk_ins(t1, c1, t2, c2, o1, q1);
      topk_ins(t1, c1, t2, c2, o2, q2);
    }
    { // off = 4
      float o1 = swz_xor_f<4>(t1); int q1 = swz_xor_i<4>(c1);
      float o2 = swz_xor_f<4>(t2); int q2 = swz_xor_i<4>(c2);
      topk_ins(t1, c1, t2, c2, o1, q1);
      topk_ins(t1, c1, t2, c2, o2, q2);
    }
    { // off = 8
      float o1 = swz_xor_f<8>(t1); int q1 = swz_xor_i<8>(c1);
      float o2 = swz_xor_f<8>(t2); int q2 = swz_xor_i<8>(c2);
      topk_ins(t1, c1, t2, c2, o1, q1);
      topk_ins(t1, c1, t2, c2, o2, q2);
    }

    if (lo == 0) {
      ws_prob[2 * row]     = t1;
      ws_prob[2 * row + 1] = t2;
      ws_idx[2 * row]      = c1;
      ws_idx[2 * row + 1]  = c2;
    }
  }

  atomicAdd(&smSums[lo],      ps0);
  atomicAdd(&smSums[lo + 16], ps1);
  atomicAdd(&smSums[lo + 32], ps2);
  atomicAdd(&smSums[lo + 48], ps3);
  __syncthreads();
  if (tid < EE) atomicAdd(&expert_sum[tid], smSums[tid]);
}

// ---------------------------------------------------------------------------
// Kernel 2: per-block, per-expert histograms of top-k expert ids.
// ---------------------------------------------------------------------------
__global__ void moe_hist(const int* __restrict__ ws_idx,
                         int* __restrict__ block_hist) {
  __shared__ int h[EE];
  const int tid = threadIdx.x, b = blockIdx.x;
  if (tid < EE) h[tid] = 0;
  __syncthreads();
  atomicAdd(&h[ws_idx[b * SLOTS_PER_BLK + tid]], 1);
  __syncthreads();
  if (tid < EE) block_hist[b * EE + tid] = h[tid];
}

// ---------------------------------------------------------------------------
// Kernel 3: exclusive scan of block histograms per expert + LB loss.
// Single block, 64 threads (one per expert).
// ---------------------------------------------------------------------------
__global__ void moe_scan_loss(const int* __restrict__ block_hist,
                              int* __restrict__ block_off,
                              const float* __restrict__ expert_sum,
                              float* __restrict__ out_loss) {
  __shared__ float part[EE];
  const int e = threadIdx.x;
  int run = 0;
  for (int b = 0; b < HBLK; ++b) {
    block_off[b * EE + e] = run;
    run += block_hist[b * EE + e];
  }
  float frac = (float)run / (float)NSLOT;          // expert_fraction
  float mean = expert_sum[e] / (float)NTOK;        // gate_mean
  part[e] = mean * frac;
  __syncthreads();
  if (e == 0) {
    float s = 0.0f;
    for (int i = 0; i < EE; ++i) s += part[i];
    out_loss[0] = s * (float)EE * 0.01f;
  }
}

// ---------------------------------------------------------------------------
// Kernel 4: in-order rank assignment within each block + normalization.
// ---------------------------------------------------------------------------
__global__ void moe_finalize(const float* __restrict__ ws_prob,
                             const int* __restrict__ ws_idx,
                             const int* __restrict__ block_off,
                             float* __restrict__ out_idx,
                             float* __restrict__ out_norm,
                             float* __restrict__ out_mask) {
  __shared__ int cnt[EE];
  __shared__ int rnk[SLOTS_PER_BLK];
  const int tid = threadIdx.x, b = blockIdx.x;
  const int s0 = b * SLOTS_PER_BLK;
  if (tid < EE) cnt[tid] = 0;
  __syncthreads();
  if (tid == 0) {
    // exact flattened-order semantics: walk slots sequentially
    for (int i = 0; i < SLOTS_PER_BLK; ++i) {
      int e = ws_idx[s0 + i];
      rnk[i] = block_off[b * EE + e] + cnt[e];
      cnt[e]++;
    }
  }
  __syncthreads();
  const int slot = s0 + tid;
  out_mask[slot] = (rnk[tid] < CAPACITY) ? 1.0f : 0.0f;
  out_idx[slot]  = (float)ws_idx[slot];
  if (tid < SLOTS_PER_BLK / 2) {
    int i0 = 2 * tid, i1 = i0 + 1;
    float mk0 = (rnk[i0] < CAPACITY) ? 1.0f : 0.0f;
    float mk1 = (rnk[i1] < CAPACITY) ? 1.0f : 0.0f;
    float q0 = ws_prob[s0 + i0] * mk0;
    float q1 = ws_prob[s0 + i1] * mk1;
    float den = fmaxf(q0 + q1, 1e-8f);
    out_norm[s0 + i0] = q0 / den;
    out_norm[s0 + i1] = q1 / den;
  }
}

// ---------------------------------------------------------------------------
// Host-side launcher
// ---------------------------------------------------------------------------
extern "C" void kernel_launch(void* const* d_in, const int* in_sizes, int n_in,
                              void* d_out, int out_size, void* d_ws,
                              size_t ws_size, hipStream_t stream) {
  (void)in_sizes; (void)n_in; (void)out_size; (void)ws_size;
  const float* hs = (const float*)d_in[0];   // [4,8192,1024] f32
  const float* wg = (const float*)d_in[1];   // [64,1024] f32

  // Output layout (flat, return order):
  //   top_k_indices[65536] | normalized_probs[65536] | gate_probs[2097152]
  //   | loss[1] | capacity_mask[65536]
  float* out      = (float*)d_out;
  float* out_idx  = out;
  float* out_norm = out + NSLOT;
  float* out_gate = out + 2 * NSLOT;
  float* out_loss = out + 2 * NSLOT + (size_t)NTOK * EE;
  float* out_mask = out_loss + 1;

  // Workspace layout
  char* ws = (char*)d_ws;
  float* ws_prob    = (float*)ws;                               // 65536 f32
  int*   ws_idx     = (int*)(ws + (size_t)NSLOT * 4);           // 65536 i32
  float* expert_sum = (float*)(ws + (size_t)NSLOT * 8);         // 64 f32
  int*   block_hist = (int*)(ws + (size_t)NSLOT * 8 + 256);     // 256*64 i32
  int*   block_off  = block_hist + HBLK * EE;                   // 256*64 i32

  moe_init<<<1, EE, 0, stream>>>(expert_sum);

  const size_t shmem = 256 + (size_t)EE * LDSB * sizeof(_Float16);
  moe_gate_gemm<<<NTOK / 128, 256, shmem, stream>>>(
      hs, wg, out_gate, ws_prob, ws_idx, expert_sum);

  moe_hist<<<HBLK, SLOTS_PER_BLK, 0, stream>>>(ws_idx, block_hist);

  moe_scan_loss<<<1, EE, 0, stream>>>(block_hist, block_off, expert_sum,
                                      out_loss);

  moe_finalize<<<HBLK, SLOTS_PER_BLK, 0, stream>>>(
      ws_prob, ws_idx, block_off, out_idx, out_norm, out_mask);
}
